// TinyRecursiveVerifier_48524540510444
// MI455X (gfx1250) — matrix operand order repaired and checked by reading, not compiled, not run on hardware
//
#include <hip/hip_runtime.h>
#include <hip/hip_bf16.h>

// ---------------------------------------------------------------------------
// TinyRecursiveVerifier for MI455X (gfx1250)
//
// Each row of z iterates independently -> one workgroup owns a 32-row tile
// and runs all 32 fixed-point steps with z resident in LDS. GEMMs use
// V_WMMA_F32_16X16X32_BF16 (bf16 in, f32 accumulate) with 2x2 register
// blocking (4 independent accumulator chains per wave) so loads overlap
// matrix ops. Weights are pre-converted to bf16 and stored FRAGMENT-MAJOR
// (tile(nt,kc) = 32 lanes x 32B contiguous) so each B-fragment is one
// coalesced 1KB wave burst from the L2-resident 8MB weight set.
// ---------------------------------------------------------------------------

typedef __attribute__((ext_vector_type(16))) __bf16 bf16x16;
typedef __attribute__((ext_vector_type(8)))  float  f32x8;

#define B_TOTAL 4096
#define D_DIM   512
#define H_DIM   2048
#define M_ROWS  32          // rows per workgroup
#define NSTEPS  32
#define EPS_CONV 0.01f
#define LN_EPS   1e-5f

// padded LDS row strides (elements): stride % 256B == 16B -> 4-bank rotation
#define XLD (D_DIM + 8)     // bf16 rows of LN output
#define HLD (H_DIM + 8)     // bf16 rows of GELU output

#define KC_X (D_DIM / 32)   // 16 k-tiles for expand GEMMs  (K=512)
#define KC_H (H_DIM / 32)   // 64 k-tiles for contract GEMMs (K=2048)

// ---------------------------------------------------------------------------
// Weight prep: fp32 -> bf16, fragment-major tiled layout.
// Tiled element (tile t, lane l, j) with t = nt*KC + kc holds
//   w[k = kc*32 + (j<8 ? (l>>4)*8 + j : 16 + (l>>4)*8 + (j-8))][n = nt*16 + (l&15)]
// so a B-fragment load is lane-contiguous: base + (t*32 + lane)*16.
//   w1,w3: src [D][H] (k-major stride H), KC=16, NT=128
//   w2,w4: src [H][D] (k-major stride D), KC=64, NT=32
// ---------------------------------------------------------------------------
__global__ __launch_bounds__(256)
void trv_prep_weights(const float* __restrict__ w1, const float* __restrict__ w2,
                      const float* __restrict__ w3, const float* __restrict__ w4,
                      __bf16* __restrict__ w1t, __bf16* __restrict__ w2t,
                      __bf16* __restrict__ w3t, __bf16* __restrict__ w4t) {
    int idx = blockIdx.x * blockDim.x + threadIdx.x;   // 0 .. D*H-1
    int j = idx & 15;
    int l = (idx >> 4) & 31;
    int t = idx >> 9;
    int krow = (j < 8) ? (l >> 4) * 8 + j : 16 + (l >> 4) * 8 + (j - 8);
    int ncol = l & 15;
    // expand weights: KC=16
    {
        int kc = t & (KC_X - 1), nt = t >> 4;
        int k = kc * 32 + krow, n = nt * 16 + ncol;
        w1t[idx] = (__bf16)w1[k * H_DIM + n];
        w3t[idx] = (__bf16)w3[k * H_DIM + n];
    }
    // contract weights: KC=64
    {
        int kc = t & (KC_H - 1), nt = t >> 6;
        int k = kc * 32 + krow, n = nt * 16 + ncol;
        w2t[idx] = (__bf16)w2[k * D_DIM + n];
        w4t[idx] = (__bf16)w4[k * D_DIM + n];
    }
}

// ---------------------------------------------------------------------------
// Fragment loaders (wave32 layouts per CDNA5 ISA 7.12.2)
// ---------------------------------------------------------------------------
// A (16x32 bf16) from LDS: lane L -> M=L&15, K = (L>>4)*8 + {0..7, 16..23}
__device__ __forceinline__ bf16x16 load_frag_A(const __bf16* base, int ld, int lane) {
    const int lo = lane & 15, hi = lane >> 4;
    const __bf16* p = base + lo * ld + hi * 8;
    bf16x16 a;
#pragma unroll
    for (int i = 0; i < 8; ++i) a[i] = p[i];
#pragma unroll
    for (int i = 0; i < 8; ++i) a[8 + i] = p[16 + i];
    return a;
}

// B (32x16 bf16) from fragment-major weights: one 32B contiguous chunk/lane,
// a full wave reads one contiguous 1KB tile.
__device__ __forceinline__ bf16x16 load_frag_Bt(const __bf16* wt, int tile, int lane) {
    const __bf16* p = wt + (((size_t)tile * 32 + lane) << 4);
    bf16x16 b;
#pragma unroll
    for (int i = 0; i < 16; ++i) b[i] = p[i];
    return b;
}

__device__ __forceinline__ f32x8 wmma_bf16(bf16x16 a, bf16x16 b, f32x8 c) {
    return __builtin_amdgcn_wmma_f32_16x16x32_bf16(
        false, a, false, b, (short)0, c, false, false);
}

__device__ __forceinline__ float wave_sum(float v) {
#pragma unroll
    for (int m = 16; m >= 1; m >>= 1) v += __shfl_xor(v, m, 32);
    return v;
}

__device__ __forceinline__ float gelu_exact(float x) {
    return 0.5f * x * (1.0f + erff(x * 0.70710678118654752f));
}

// LayerNorm, 4 rows per wave: src fp32 [M_ROWS][D] -> dst bf16 [M_ROWS][XLD]
__device__ __forceinline__ void layernorm_rows(const float* src, __bf16* dst,
                                               const float* __restrict__ g,
                                               const float* __restrict__ bta,
                                               int wave, int lane) {
#pragma unroll
    for (int rr = 0; rr < M_ROWS / 8; ++rr) {
        int r = wave * (M_ROWS / 8) + rr;
        float s = 0.f, q = 0.f;
#pragma unroll
        for (int j = 0; j < D_DIM / 32; ++j) {
            float v = src[r * D_DIM + lane + 32 * j];
            s += v; q += v * v;
        }
        s = wave_sum(s); q = wave_sum(q);
        float m  = s * (1.0f / D_DIM);
        float vr = q * (1.0f / D_DIM) - m * m;
        float rs = rsqrtf(vr + LN_EPS);
#pragma unroll
        for (int j = 0; j < D_DIM / 32; ++j) {
            int c = lane + 32 * j;
            dst[r * XLD + c] = (__bf16)((src[r * D_DIM + c] - m) * rs * g[c] + bta[c]);
        }
    }
}

// ---------------------------------------------------------------------------
// GEMM expand: [32 x 512] (bf16 LDS) x W[512 x 2048] -> GELU -> hb bf16.
// 2x2 supertiles: both M-tiles x an N-tile pair; 4 accumulators, B prefetch.
// 64 nt-pairs -> 8 supertiles per wave.
// ---------------------------------------------------------------------------
__device__ __forceinline__ void gemm_expand(const __bf16* xb, const __bf16* __restrict__ wt,
                                            const float* __restrict__ bias, __bf16* hb,
                                            int wave, int lane) {
    const int lo = lane & 15, hi = lane >> 4;
    for (int st = wave; st < (H_DIM / 32); st += 8) {
        const int nt0 = st * 2, nt1 = st * 2 + 1;
        f32x8 acc00 = {}, acc01 = {}, acc10 = {}, acc11 = {};
        bf16x16 b0 = load_frag_Bt(wt, nt0 * KC_X, lane);
        bf16x16 b1 = load_frag_Bt(wt, nt1 * KC_X, lane);
#pragma unroll 4
        for (int kc = 0; kc < KC_X; ++kc) {
            bf16x16 a0 = load_frag_A(xb + kc * 32, XLD, lane);
            bf16x16 a1 = load_frag_A(xb + 16 * XLD + kc * 32, XLD, lane);
            int kn = (kc + 1 < KC_X) ? kc + 1 : kc;
            bf16x16 nb0 = load_frag_Bt(wt, nt0 * KC_X + kn, lane);
            bf16x16 nb1 = load_frag_Bt(wt, nt1 * KC_X + kn, lane);
            acc00 = wmma_bf16(a0, b0, acc00);
            acc01 = wmma_bf16(a0, b1, acc01);
            acc10 = wmma_bf16(a1, b0, acc10);
            acc11 = wmma_bf16(a1, b1, acc11);
            b0 = nb0; b1 = nb1;
        }
        const int col0 = nt0 * 16 + lo, col1 = nt1 * 16 + lo;
        const float bv0 = bias[col0], bv1 = bias[col1];
        const int r0 = hi * 8;
#pragma unroll
        for (int r = 0; r < 8; ++r) {
            hb[(r0 + r) * HLD + col0]      = (__bf16)gelu_exact(acc00[r] + bv0);
            hb[(r0 + r) * HLD + col1]      = (__bf16)gelu_exact(acc01[r] + bv1);
            hb[(16 + r0 + r) * HLD + col0] = (__bf16)gelu_exact(acc10[r] + bv0);
            hb[(16 + r0 + r) * HLD + col1] = (__bf16)gelu_exact(acc11[r] + bv1);
        }
    }
}

// ---------------------------------------------------------------------------
// GEMM contract: [32 x 2048] (bf16 LDS) x W[2048 x 512] -> +bias -> fp32 tf.
// 16 nt-pairs -> 2 supertiles per wave, K = 64 tiles.
// ---------------------------------------------------------------------------
__device__ __forceinline__ void gemm_contract(const __bf16* hb, const __bf16* __restrict__ wt,
                                              const float* __restrict__ bias, float* tf,
                                              int wave, int lane) {
    const int lo = lane & 15, hi = lane >> 4;
    for (int st = wave; st < (D_DIM / 32); st += 8) {
        const int nt0 = st * 2, nt1 = st * 2 + 1;
        f32x8 acc00 = {}, acc01 = {}, acc10 = {}, acc11 = {};
        bf16x16 b0 = load_frag_Bt(wt, nt0 * KC_H, lane);
        bf16x16 b1 = load_frag_Bt(wt, nt1 * KC_H, lane);
#pragma unroll 4
        for (int kc = 0; kc < KC_H; ++kc) {
            bf16x16 a0 = load_frag_A(hb + kc * 32, HLD, lane);
            bf16x16 a1 = load_frag_A(hb + 16 * HLD + kc * 32, HLD, lane);
            int kn = (kc + 1 < KC_H) ? kc + 1 : kc;
            bf16x16 nb0 = load_frag_Bt(wt, nt0 * KC_H + kn, lane);
            bf16x16 nb1 = load_frag_Bt(wt, nt1 * KC_H + kn, lane);
            acc00 = wmma_bf16(a0, b0, acc00);
            acc01 = wmma_bf16(a0, b1, acc01);
            acc10 = wmma_bf16(a1, b0, acc10);
            acc11 = wmma_bf16(a1, b1, acc11);
            b0 = nb0; b1 = nb1;
        }
        const int col0 = nt0 * 16 + lo, col1 = nt1 * 16 + lo;
        const float bv0 = bias[col0], bv1 = bias[col1];
        const int r0 = hi * 8;
#pragma unroll
        for (int r = 0; r < 8; ++r) {
            tf[(r0 + r) * D_DIM + col0]      = acc00[r] + bv0;
            tf[(r0 + r) * D_DIM + col1]      = acc01[r] + bv1;
            tf[(16 + r0 + r) * D_DIM + col0] = acc10[r] + bv0;
            tf[(16 + r0 + r) * D_DIM + col1] = acc11[r] + bv1;
        }
    }
}

// ---------------------------------------------------------------------------
// Main kernel: 128 blocks x 256 threads (8 waves); full 32-step fixed point
// for a 32-row tile with z resident in LDS.
// ---------------------------------------------------------------------------
__global__ __launch_bounds__(256)
void trv_main(const float* __restrict__ z_init,
              const float* __restrict__ ln1_g, const float* __restrict__ ln1_b,
              const float* __restrict__ b1,    const float* __restrict__ b2,
              const float* __restrict__ ln2_g, const float* __restrict__ ln2_b,
              const float* __restrict__ b3,    const float* __restrict__ b4,
              const float* __restrict__ clf_w, const float* __restrict__ clf_b,
              const __bf16* __restrict__ w1t,  const __bf16* __restrict__ w2t,
              const __bf16* __restrict__ w3t,  const __bf16* __restrict__ w4t,
              float* __restrict__ out_logits, float* __restrict__ out_mask,
              float* __restrict__ out_z) {
    extern __shared__ char smem[];
    float*  zf = (float*)smem;                    // [M_ROWS][D]  fp32 state
    float*  tf = zf + M_ROWS * D_DIM;             // [M_ROWS][D]  fp32 temp
    __bf16* xb = (__bf16*)(tf + M_ROWS * D_DIM);  // [M_ROWS][XLD] LN out
    __bf16* hb = xb + M_ROWS * XLD;               // [M_ROWS][HLD] GELU out
    int*    mk = (int*)(hb + M_ROWS * HLD);       // [M_ROWS] converged flags

    const int tid  = threadIdx.x;
    const int lane = tid & 31;
    const int wave = tid >> 5;                    // 0..7
    const int rowbase = blockIdx.x * M_ROWS;

    for (int i = tid; i < M_ROWS * D_DIM; i += 256)
        zf[i] = z_init[(size_t)rowbase * D_DIM + i];
    if (tid < M_ROWS) mk[tid] = 0;
    __syncthreads();

    for (int step = 0; step < NSTEPS; ++step) {
        layernorm_rows(zf, xb, ln1_g, ln1_b, wave, lane);
        __syncthreads();
        gemm_expand(xb, w1t, b1, hb, wave, lane);
        __syncthreads();
        gemm_contract(hb, w2t, b2, tf, wave, lane);
        __syncthreads();
        layernorm_rows(tf, xb, ln2_g, ln2_b, wave, lane);
        __syncthreads();
        gemm_expand(xb, w3t, b3, hb, wave, lane);
        __syncthreads();
        gemm_contract(hb, w4t, b4, tf, wave, lane);   // tf = delta
        __syncthreads();

        // ||delta||, mask-first-then-update (matches reference ordering)
#pragma unroll
        for (int rr = 0; rr < M_ROWS / 8; ++rr) {
            int r = wave * (M_ROWS / 8) + rr;
            float q = 0.f;
#pragma unroll
            for (int j = 0; j < D_DIM / 32; ++j) {
                float v = tf[r * D_DIM + lane + 32 * j];
                q += v * v;
            }
            q = wave_sum(q);
            int conv = mk[r] | ((sqrtf(q) < EPS_CONV) ? 1 : 0);
            if (lane == 0) mk[r] = conv;
            if (!conv) {
#pragma unroll
                for (int j = 0; j < D_DIM / 32; ++j) {
                    int c = lane + 32 * j;
                    zf[r * D_DIM + c] += tf[r * D_DIM + c];
                }
            }
        }
        __syncthreads();
    }

    // logits = z @ clf_w + clf_b ; outputs in return order: logits, mask, z
#pragma unroll
    for (int rr = 0; rr < M_ROWS / 8; ++rr) {
        int r = wave * (M_ROWS / 8) + rr;
        float s = 0.f;
#pragma unroll
        for (int j = 0; j < D_DIM / 32; ++j) {
            int c = lane + 32 * j;
            s += zf[r * D_DIM + c] * clf_w[c];
        }
        s = wave_sum(s);
        if (lane == 0) {
            out_logits[rowbase + r] = s + clf_b[0];
            out_mask[rowbase + r]   = mk[r] ? 1.0f : 0.0f;
        }
    }
    for (int i = tid; i < M_ROWS * D_DIM; i += 256)
        out_z[(size_t)rowbase * D_DIM + i] = zf[i];
}

// ---------------------------------------------------------------------------
extern "C" void kernel_launch(void* const* d_in, const int* in_sizes, int n_in,
                              void* d_out, int out_size, void* d_ws, size_t ws_size,
                              hipStream_t stream) {
    (void)in_sizes; (void)n_in; (void)out_size; (void)ws_size;
    const float* z_init = (const float*)d_in[0];
    const float* ln1_g  = (const float*)d_in[1];
    const float* ln1_b  = (const float*)d_in[2];
    const float* w1     = (const float*)d_in[3];
    const float* b1     = (const float*)d_in[4];
    const float* w2     = (const float*)d_in[5];
    const float* b2     = (const float*)d_in[6];
    const float* ln2_g  = (const float*)d_in[7];
    const float* ln2_b  = (const float*)d_in[8];
    const float* w3     = (const float*)d_in[9];
    const float* b3     = (const float*)d_in[10];
    const float* w4     = (const float*)d_in[11];
    const float* b4     = (const float*)d_in[12];
    const float* clf_w  = (const float*)d_in[13];
    const float* clf_b  = (const float*)d_in[14];

    const size_t WELEMS = (size_t)D_DIM * H_DIM;   // 1,048,576 per matrix
    __bf16* w1t = (__bf16*)d_ws;
    __bf16* w2t = w1t + WELEMS;
    __bf16* w3t = w2t + WELEMS;
    __bf16* w4t = w3t + WELEMS;

    float* out_logits = (float*)d_out;             // [4096]
    float* out_mask   = out_logits + B_TOTAL;      // [4096]
    float* out_z      = out_mask + B_TOTAL;        // [4096*512]

    trv_prep_weights<<<(int)(WELEMS / 256), 256, 0, stream>>>(
        w1, w2, w3, w4, w1t, w2t, w3t, w4t);

    size_t lds_bytes = (size_t)M_ROWS * D_DIM * 4 * 2       // zf + tf
                     + (size_t)M_ROWS * XLD * 2             // xb
                     + (size_t)M_ROWS * HLD * 2             // hb
                     + (size_t)M_ROWS * 4;                  // mk
    trv_main<<<B_TOTAL / M_ROWS, 256, lds_bytes, stream>>>(
        z_init, ln1_g, ln1_b, b1, b2, ln2_g, ln2_b, b3, b4,
        clf_w, clf_b, w1t, w2t, w3t, w4t,
        out_logits, out_mask, out_z);
}
